// BERT_FOL_T_41266045780243
// MI455X (gfx1250) — matrix operand order repaired and checked by reading, not compiled, not run on hardware
//
#include <hip/hip_runtime.h>
#include <hip/hip_bf16.h>

// ---------------- problem constants ----------------
#define BB 32
#define SS 128
#define HH 768
#define LL 12
#define NHH 12
#define FFF 3072
#define DD 64
#define BS (BB * SS)          // 4096 token rows

// ---------------- vector types ----------------
typedef __attribute__((ext_vector_type(16))) __bf16          v16bf;
typedef __attribute__((ext_vector_type(8)))  float           v8f;
typedef __attribute__((ext_vector_type(8)))  unsigned short  us8;
typedef __attribute__((ext_vector_type(4)))  int             v4i;

union Frag {                  // fragment assembled from two 16B LDS reads
    us8   h[2];
    v16bf b;
};
union U8 {                    // pack 8 bf16 from 4 u32
    unsigned int u32[4];
    us8          v;
};

// ---- CDNA5 async global->LDS path (guarded; falls back to sync copy) ----
#if defined(__AMDGCN__) && __has_builtin(__builtin_amdgcn_global_load_async_to_lds_b128)
#define USE_ASYNC_LDS 1
typedef __attribute__((address_space(1))) v4i as1_v4i;   // global int4*
typedef __attribute__((address_space(3))) v4i as3_v4i;   // LDS int4*
#else
#define USE_ASYNC_LDS 0
#endif

__device__ __forceinline__ void wait_async0() {
#if defined(__AMDGCN__)
#if __has_builtin(__builtin_amdgcn_s_wait_asynccnt)
    __builtin_amdgcn_s_wait_asynccnt(0);
#elif USE_ASYNC_LDS
    asm volatile("s_wait_asynccnt 0x0" ::: "memory");
#endif
#endif
}

// fp32 -> bf16 round-to-nearest-even, raw u16
__device__ __forceinline__ unsigned short f2bf(float f) {
    unsigned int u = __float_as_uint(f);
    unsigned int r = u + 0x7FFFu + ((u >> 16) & 1u);
    return (unsigned short)(r >> 16);
}
// pack two fp32 -> (hi<<16)|lo bf16 pair
__device__ __forceinline__ unsigned int pk2bf(float lo, float hi) {
    unsigned int a = __float_as_uint(lo), b = __float_as_uint(hi);
    a = a + 0x7FFFu + ((a >> 16) & 1u);
    b = b + 0x7FFFu + ((b >> 16) & 1u);
    return (a >> 16) | (b & 0xFFFF0000u);
}

__device__ __forceinline__ float gelu_tanh(float x) {
    float x3 = x * x * x;
    return 0.5f * x * (1.0f + tanhf(0.79788456080286535588f * (x + 0.044715f * x3)));
}

// ============================================================
// Weight convert + transpose:  W[k][n] fp32  ->  Wt[n][k] bf16
// ============================================================
__global__ __launch_bounds__(256)
void convert_w_kernel(const float* __restrict__ W, unsigned short* __restrict__ Wt,
                      int K, int N) {
    int gid = blockIdx.x * 256 + threadIdx.x;
    if (gid >= N * (K >> 3)) return;
    int n  = gid % N;
    int kc = (gid / N) * 8;
    U8 u;
#pragma unroll
    for (int j = 0; j < 4; j++) {
        float f0 = W[(size_t)(kc + 2 * j)     * N + n];
        float f1 = W[(size_t)(kc + 2 * j + 1) * N + n];
        u.u32[j] = pk2bf(f0, f1);
    }
    *(us8*)(Wt + (size_t)n * K + kc) = u.v;
}

// ============================================================
// Embedding gather + LayerNorm.  One wave per token row.
// ============================================================
__global__ __launch_bounds__(256)
void embed_ln_kernel(const int* __restrict__ idx, const int* __restrict__ typ,
                     const float* __restrict__ wemb, const float* __restrict__ pemb,
                     const float* __restrict__ temb,
                     const float* __restrict__ g, const float* __restrict__ bb,
                     float* __restrict__ x, unsigned short* __restrict__ xb) {
    int wave = (blockIdx.x * blockDim.x + threadIdx.x) >> 5;
    int lane = threadIdx.x & 31;
    if (wave >= BS) return;
    int s = wave % SS;
    const float* wr = wemb + (size_t)idx[wave] * HH;
    const float* tr = temb + (size_t)typ[wave] * HH;
    const float* pr = pemb + (size_t)s * HH;

    float vv[HH / 32];
    float sum = 0.f;
#pragma unroll
    for (int i = 0; i < HH / 32; i++) {
        int c = lane + 32 * i;
        float t = wr[c] + pr[c] + tr[c];
        vv[i] = t; sum += t;
    }
#pragma unroll
    for (int m = 16; m >= 1; m >>= 1) sum += __shfl_xor(sum, m, 32);
    float mu = sum * (1.0f / HH);
    float sq = 0.f;
#pragma unroll
    for (int i = 0; i < HH / 32; i++) { float d = vv[i] - mu; sq += d * d; }
#pragma unroll
    for (int m = 16; m >= 1; m >>= 1) sq += __shfl_xor(sq, m, 32);
    float inv = rsqrtf(sq * (1.0f / HH) + 1e-12f);
#pragma unroll
    for (int i = 0; i < HH / 32; i++) {
        int c = lane + 32 * i;
        float o = g[c] * (vv[i] - mu) * inv + bb[c];
        x[(size_t)wave * HH + c]  = o;
        xb[(size_t)wave * HH + c] = f2bf(o);
    }
}

// ============================================================
// Residual add + LayerNorm (in place on x) + bf16 mirror.
// ============================================================
__global__ __launch_bounds__(256)
void add_ln_kernel(float* __restrict__ x, const float* __restrict__ delta,
                   const float* __restrict__ g, const float* __restrict__ bb,
                   unsigned short* __restrict__ xb) {
    int wave = (blockIdx.x * blockDim.x + threadIdx.x) >> 5;
    int lane = threadIdx.x & 31;
    if (wave >= BS) return;
    float vv[HH / 32];
    float sum = 0.f;
#pragma unroll
    for (int i = 0; i < HH / 32; i++) {
        int c = lane + 32 * i;
        float t = x[(size_t)wave * HH + c] + delta[(size_t)wave * HH + c];
        vv[i] = t; sum += t;
    }
#pragma unroll
    for (int m = 16; m >= 1; m >>= 1) sum += __shfl_xor(sum, m, 32);
    float mu = sum * (1.0f / HH);
    float sq = 0.f;
#pragma unroll
    for (int i = 0; i < HH / 32; i++) { float d = vv[i] - mu; sq += d * d; }
#pragma unroll
    for (int m = 16; m >= 1; m >>= 1) sq += __shfl_xor(sq, m, 32);
    float inv = rsqrtf(sq * (1.0f / HH) + 1e-12f);
#pragma unroll
    for (int i = 0; i < HH / 32; i++) {
        int c = lane + 32 * i;
        float o = g[c] * (vv[i] - mu) * inv + bb[c];
        x[(size_t)wave * HH + c]  = o;
        xb[(size_t)wave * HH + c] = f2bf(o);
    }
}

// ============================================================
// WMMA GEMM, all-bf16 operands, double-buffered LDS, async staging:
//   C[M,N] = act(A[M,K] @ Wt[n][k]^T + bias[N])
// Macro tile 128x128, 8 waves, each wave 32(M) x 64(N) = 2x4 WMMA
// accumulators, K stepped by 32.
// flags: bit0 = GELU epilogue, bit1 = bf16 output.
// ============================================================
__global__ __launch_bounds__(256)
void gemm_bf16_kernel(const unsigned short* __restrict__ A,
                      const unsigned short* __restrict__ Wt,
                      const float* __restrict__ bias, void* __restrict__ Cout,
                      int M, int N, int K, int flags) {
    __shared__ alignas(16) unsigned short As[2][128][40];   // [buf][m][k] pad 8
    __shared__ alignas(16) unsigned short Bs[2][128][40];   // [buf][n][k] pad 8

    int tid  = threadIdx.x;
    int wave = tid >> 5;
    int lane = tid & 31;
    int ll = lane & 15, lh = lane >> 4;
    int m0 = blockIdx.x * 128;
    int n0 = blockIdx.y * 128;
    int wm = (wave & 3) * 32;
    int wn = (wave >> 2) * 64;

    // staging: each thread copies 16 bf16 of A and 16 of W per K step
    int sr = tid >> 1;               // 0..127
    int sc = (tid & 1) * 16;         // 0 or 16
    int am = m0 + sr; if (am >= M) am = M - 1;     // clamp: extra rows unstored
    const unsigned short* aSrc = A  + (size_t)am * K + sc;
    const unsigned short* bSrc = Wt + (size_t)(n0 + sr) * K + sc;

    auto stage = [&](int p, int k0) {
#if USE_ASYNC_LDS
        __builtin_amdgcn_global_load_async_to_lds_b128(
            (as1_v4i*)(aSrc + k0),     (as3_v4i*)&As[p][sr][sc],     0, 0);
        __builtin_amdgcn_global_load_async_to_lds_b128(
            (as1_v4i*)(aSrc + k0 + 8), (as3_v4i*)&As[p][sr][sc + 8], 0, 0);
        __builtin_amdgcn_global_load_async_to_lds_b128(
            (as1_v4i*)(bSrc + k0),     (as3_v4i*)&Bs[p][sr][sc],     0, 0);
        __builtin_amdgcn_global_load_async_to_lds_b128(
            (as1_v4i*)(bSrc + k0 + 8), (as3_v4i*)&Bs[p][sr][sc + 8], 0, 0);
#else
        *(us8*)&As[p][sr][sc]     = *(const us8*)(aSrc + k0);
        *(us8*)&As[p][sr][sc + 8] = *(const us8*)(aSrc + k0 + 8);
        *(us8*)&Bs[p][sr][sc]     = *(const us8*)(bSrc + k0);
        *(us8*)&Bs[p][sr][sc + 8] = *(const us8*)(bSrc + k0 + 8);
        if (k0 + 32 < K) {
            __builtin_prefetch(aSrc + k0 + 32, 0, 1);
            __builtin_prefetch(bSrc + k0 + 32, 0, 1);
        }
#endif
    };

    v8f acc[2][4] = {};

    stage(0, 0);
    int p = 0;
    for (int k0 = 0; k0 < K; k0 += 32) {
        wait_async0();          // tile p complete (no-op on sync path)
        __syncthreads();        // includes dscnt-0: prior reads/writes retired
        if (k0 + 32 < K) stage(p ^ 1, k0 + 32);   // overlap next tile with WMMAs

        Frag af[2], bf[4];
#pragma unroll
        for (int mt = 0; mt < 2; mt++) {
            const us8* q = (const us8*)&As[p][wm + 16 * mt + ll][8 * lh];
            af[mt].h[0] = q[0];      // K = 8*lh + 0..7
            af[mt].h[1] = q[2];      // K = 8*lh + 16..23
        }
#pragma unroll
        for (int nt = 0; nt < 4; nt++) {
            const us8* q = (const us8*)&Bs[p][wn + 16 * nt + ll][8 * lh];
            bf[nt].h[0] = q[0];      // K = 8*lh + 0..7
            bf[nt].h[1] = q[2];      // K = 8*lh + 16..23
        }
#pragma unroll
        for (int mt = 0; mt < 2; mt++)
#pragma unroll
            for (int nt = 0; nt < 4; nt++)
                acc[mt][nt] = __builtin_amdgcn_wmma_f32_16x16x32_bf16(
                    false, af[mt].b, false, bf[nt].b,
                    (short)0, acc[mt][nt], false, false);
        p ^= 1;
    }

    // ---- epilogue ----
    float*          Cf = (float*)Cout;
    unsigned short* Cb = (unsigned short*)Cout;
#pragma unroll
    for (int mt = 0; mt < 2; mt++)
#pragma unroll
        for (int nt = 0; nt < 4; nt++)
#pragma unroll
            for (int r = 0; r < 8; r++) {
                int gm = m0 + wm + 16 * mt + r + 8 * lh;
                int gn = n0 + wn + 16 * nt + ll;
                if (gm < M) {
                    float v = acc[mt][nt][r] + bias[gn];
                    if (flags & 1) v = gelu_tanh(v);
                    if (flags & 2) Cb[(size_t)gm * N + gn] = f2bf(v);
                    else           Cf[(size_t)gm * N + gn] = v;
                }
            }
}

// ============================================================
// Attention: one workgroup per (b, h); 8 waves x 16 queries.
// Q/K/V arrive as bf16; scores/softmax fp32; ctx written bf16.
// ============================================================
struct AttnLdsA { unsigned short Ks[128][72]; };                              // [key][d]
struct AttnLdsB { unsigned short Vt[64][136]; unsigned short Ps[128][136]; }; // [d][key],[q][key]
union  AttnLds  { AttnLdsA a; AttnLdsB b; };

__global__ __launch_bounds__(256)
void attention_kernel(const unsigned short* __restrict__ q,
                      const unsigned short* __restrict__ k,
                      const unsigned short* __restrict__ v,
                      const int* __restrict__ mask,
                      unsigned short* __restrict__ ctx) {
    __shared__ alignas(16) AttnLds lds;

    int b = blockIdx.x / NHH;
    int h = blockIdx.x % NHH;
    int tid = threadIdx.x;
    int wave = tid >> 5;
    int lane = tid & 31;
    int ll = lane & 15, lh = lane >> 4;

    // ---- stage K tile [128 keys][64 d] : async b128 copies ----
#pragma unroll
    for (int c = tid; c < 1024; c += 256) {
        int key = c >> 3, dc = (c & 7) * 8;
#if USE_ASYNC_LDS
        __builtin_amdgcn_global_load_async_to_lds_b128(
            (as1_v4i*)(k + (size_t)(b * SS + key) * HH + h * DD + dc),
            (as3_v4i*)&lds.a.Ks[key][dc], 0, 0);
#else
        *(us8*)&lds.a.Ks[key][dc] =
            *(const us8*)(k + (size_t)(b * SS + key) * HH + h * DD + dc);
#endif
    }

    // ---- Q fragments: two b128 global loads each ----
    const unsigned short* qp = q + (size_t)(b * SS + wave * 16 + ll) * HH + h * DD;
    Frag aq[2];
#pragma unroll
    for (int ds = 0; ds < 2; ds++) {
        int db = 32 * ds + 8 * lh;
        aq[ds].h[0] = *(const us8*)(qp + db);
        aq[ds].h[1] = *(const us8*)(qp + db + 16);
    }
    wait_async0();
    __syncthreads();

    // ---- scores: 8 key-tiles, d-loop of 2 ----
    v8f sacc[8] = {};
#pragma unroll
    for (int nt = 0; nt < 8; nt++) {
#pragma unroll
        for (int ds = 0; ds < 2; ds++) {
            Frag kf;
            const us8* p = (const us8*)&lds.a.Ks[16 * nt + ll][32 * ds + 8 * lh];
            kf.h[0] = p[0];
            kf.h[1] = p[2];
            sacc[nt] = __builtin_amdgcn_wmma_f32_16x16x32_bf16(
                false, aq[ds].b, false, kf.b, (short)0, sacc[nt], false, false);
        }
    }
    __syncthreads();   // all waves done with Ks before LdsB overwrite

    // ---- stage V transposed [d][key] ----
#pragma unroll
    for (int c = tid; c < 1024; c += 256) {
        int key = c >> 3, dc = (c & 7) * 8;
        us8 vv = *(const us8*)(v + (size_t)(b * SS + key) * HH + h * DD + dc);
#pragma unroll
        for (int j = 0; j < 8; j++) lds.b.Vt[dc + j][key] = vv[j];
    }

    // ---- softmax over 128 keys (scale + mask bias) -> P bf16 in LDS ----
    float bia[8];
#pragma unroll
    for (int nt = 0; nt < 8; nt++)
        bia[nt] = (1.0f - (float)mask[b * SS + 16 * nt + ll]) * -1e9f;
#pragma unroll
    for (int nt = 0; nt < 8; nt++)
#pragma unroll
        for (int r = 0; r < 8; r++)
            sacc[nt][r] = sacc[nt][r] * 0.125f + bia[nt];

#pragma unroll
    for (int r = 0; r < 8; r++) {
        float m = sacc[0][r];
#pragma unroll
        for (int nt = 1; nt < 8; nt++) m = fmaxf(m, sacc[nt][r]);
#pragma unroll
        for (int off = 1; off < 16; off <<= 1) m = fmaxf(m, __shfl_xor(m, off, 32));
        float s = 0.f;
#pragma unroll
        for (int nt = 0; nt < 8; nt++) {
            float e = __expf(sacc[nt][r] - m);
            sacc[nt][r] = e; s += e;
        }
#pragma unroll
        for (int off = 1; off < 16; off <<= 1) s += __shfl_xor(s, off, 32);
        float inv = 1.0f / s;
#pragma unroll
        for (int nt = 0; nt < 8; nt++)
            lds.b.Ps[wave * 16 + r + 8 * lh][16 * nt + ll] = f2bf(sacc[nt][r] * inv);
    }
    __syncthreads();

    // ---- ctx = P[16x128] @ V[128x64] ----
    v8f cacc[4] = {};
#pragma unroll
    for (int ks = 0; ks < 4; ks++) {
        Frag pa;
        const us8* pp = (const us8*)&lds.b.Ps[wave * 16 + ll][32 * ks + 8 * lh];
        pa.h[0] = pp[0];
        pa.h[1] = pp[2];
#pragma unroll
        for (int nt = 0; nt < 4; nt++) {
            Frag vf;
            const us8* vp = (const us8*)&lds.b.Vt[16 * nt + ll][32 * ks + 8 * lh];
            vf.h[0] = vp[0];
            vf.h[1] = vp[2];
            cacc[nt] = __builtin_amdgcn_wmma_f32_16x16x32_bf16(
                false, pa.b, false, vf.b, (short)0, cacc[nt], false, false);
        }
    }
#pragma unroll
    for (int nt = 0; nt < 4; nt++)
#pragma unroll
        for (int r = 0; r < 8; r++) {
            int s = wave * 16 + r + 8 * lh;
            ctx[(size_t)(b * SS + s) * HH + h * DD + 16 * nt + ll] = f2bf(cacc[nt][r]);
        }
}

// ============================================================
// Masked mean pooling -> bf16 pooled (GEMM input)
// ============================================================
__global__ __launch_bounds__(256)
void pool_kernel(const float* __restrict__ x, const int* __restrict__ idx,
                 unsigned short* __restrict__ pooledb) {
    int b = blockIdx.x;
    float c = 0.f;
    for (int s = 0; s < SS; s++) c += (idx[b * SS + s] != 0) ? 1.f : 0.f;
    float inv = 1.0f / fmaxf(c, 1.0f);
    for (int hh = threadIdx.x; hh < HH; hh += 256) {
        float acc = 0.f;
        for (int s = 0; s < SS; s++)
            if (idx[b * SS + s] != 0)
                acc += x[(size_t)(b * SS + s) * HH + hh];
        pooledb[(size_t)b * HH + hh] = f2bf(acc * inv);
    }
}

// ============================================================
// logits[b,j] = dot(out2[b,:], word_emb[pidx[j],:])   (B x 3)
// ============================================================
__global__ void logits_kernel(const float* __restrict__ out2,
                              const float* __restrict__ wemb,
                              const int* __restrict__ pidx,
                              float* __restrict__ logits) {
    int b = blockIdx.x;
    int j = threadIdx.y;
    int lane = threadIdx.x;
    const float* wr = wemb + (size_t)pidx[j] * HH;
    float s = 0.f;
    for (int c = lane; c < HH; c += 32)
        s += out2[(size_t)b * HH + c] * wr[c];
#pragma unroll
    for (int m = 16; m >= 1; m >>= 1) s += __shfl_xor(s, m, 32);
    if (lane == 0) logits[b * 3 + j] = s;
}

// ============================================================
// host launcher
// ============================================================
extern "C" void kernel_launch(void* const* d_in, const int* in_sizes, int n_in,
                              void* d_out, int out_size, void* d_ws, size_t ws_size,
                              hipStream_t stream) {
    (void)in_sizes; (void)n_in; (void)out_size; (void)ws_size;

    const int*   idx   = (const int*)d_in[0];
    const int*   typ   = (const int*)d_in[1];
    const int*   msk   = (const int*)d_in[2];
    const int*   pidx  = (const int*)d_in[3];
    const float* wemb  = (const float*)d_in[4];
    const float* pemb  = (const float*)d_in[5];
    const float* temb  = (const float*)d_in[6];
    const float* elns  = (const float*)d_in[7];
    const float* elnb  = (const float*)d_in[8];
    const float* Wq    = (const float*)d_in[9];
    const float* bq    = (const float*)d_in[10];
    const float* Wk    = (const float*)d_in[11];
    const float* bk    = (const float*)d_in[12];
    const float* Wv    = (const float*)d_in[13];
    const float* bv    = (const float*)d_in[14];
    const float* Wo    = (const float*)d_in[15];
    const float* bo    = (const float*)d_in[16];
    const float* ln1s  = (const float*)d_in[17];
    const float* ln1b  = (const float*)d_in[18];
    const float* W1    = (const float*)d_in[19];
    const float* b1    = (const float*)d_in[20];
    const float* W2    = (const float*)d_in[21];
    const float* b2    = (const float*)d_in[22];
    const float* ln2s  = (const float*)d_in[23];
    const float* ln2b  = (const float*)d_in[24];
    const float* d2W   = (const float*)d_in[25];
    const float* d2b   = (const float*)d_in[26];

    // ---------- workspace carve-up (256B aligned blocks) ----------
    char* wp = (char*)d_ws;
    auto alloc = [&](size_t bytes) -> void* {
        void* r = (void*)wp;
        wp += (bytes + 255) & ~(size_t)255;
        return r;
    };
    const size_t BSH = (size_t)BS * HH;
    const size_t BSF = (size_t)BS * FFF;
    const size_t HW  = (size_t)HH * HH;
    const size_t HF  = (size_t)HH * FFF;

    float*          x    = (float*)alloc(BSH * 4);         // fp32 hidden
    float*          tmp  = (float*)alloc(BSH * 4);         // proj / ffn-out fp32
    unsigned short* xb   = (unsigned short*)alloc(BSH * 2);
    unsigned short* qb   = (unsigned short*)alloc(BSH * 2);
    unsigned short* kb   = (unsigned short*)alloc(BSH * 2);
    unsigned short* vb   = (unsigned short*)alloc(BSH * 2);
    unsigned short* cb   = (unsigned short*)alloc(BSH * 2);
    unsigned short* fb   = (unsigned short*)alloc(BSF * 2);
    unsigned short* poob = (unsigned short*)alloc((size_t)BB * HH * 2);
    float*          out2 = (float*)alloc((size_t)BB * HH * 4);
    unsigned short* wt   = (unsigned short*)alloc((4 * HW + 2 * HF) * LL * 2);
    unsigned short* d2Wt = (unsigned short*)alloc(HW * 2);

    dim3 blk(256);

    // ---------- one-time (per launch) weight convert+transpose ----------
    const int gHW = (int)((HW / 8 + 255) / 256);
    const int gHF = (int)((HF / 8 + 255) / 256);
    for (int l = 0; l < LL; l++) {
        unsigned short* base = wt + (size_t)l * (4 * HW + 2 * HF);
        convert_w_kernel<<<gHW, blk, 0, stream>>>(Wq + (size_t)l * HW, base,          HH, HH);
        convert_w_kernel<<<gHW, blk, 0, stream>>>(Wk + (size_t)l * HW, base + HW,     HH, HH);
        convert_w_kernel<<<gHW, blk, 0, stream>>>(Wv + (size_t)l * HW, base + 2 * HW, HH, HH);
        convert_w_kernel<<<gHW, blk, 0, stream>>>(Wo + (size_t)l * HW, base + 3 * HW, HH, HH);
        convert_w_kernel<<<gHF, blk, 0, stream>>>(W1 + (size_t)l * HF, base + 4 * HW,      HH,  FFF);
        convert_w_kernel<<<gHF, blk, 0, stream>>>(W2 + (size_t)l * HF, base + 4 * HW + HF, FFF, HH);
    }
    convert_w_kernel<<<gHW, blk, 0, stream>>>(d2W, d2Wt, HH, HH);

    // ---------- embeddings + LN ----------
    embed_ln_kernel<<<BS / 8, blk, 0, stream>>>(idx, typ, wemb, pemb, temb, elns, elnb, x, xb);

    dim3 gH(BS / 128, HH / 128);     // 32 x 6
    dim3 gF(BS / 128, FFF / 128);    // 32 x 24

    for (int l = 0; l < LL; l++) {
        unsigned short* base = wt + (size_t)l * (4 * HW + 2 * HF);
        unsigned short* wqT = base;
        unsigned short* wkT = base + HW;
        unsigned short* wvT = base + 2 * HW;
        unsigned short* woT = base + 3 * HW;
        unsigned short* w1T = base + 4 * HW;
        unsigned short* w2T = base + 4 * HW + HF;

        gemm_bf16_kernel<<<gH, blk, 0, stream>>>(xb, wqT, bq + (size_t)l * HH, qb, BS, HH, HH, 2);
        gemm_bf16_kernel<<<gH, blk, 0, stream>>>(xb, wkT, bk + (size_t)l * HH, kb, BS, HH, HH, 2);
        gemm_bf16_kernel<<<gH, blk, 0, stream>>>(xb, wvT, bv + (size_t)l * HH, vb, BS, HH, HH, 2);

        attention_kernel<<<BB * NHH, blk, 0, stream>>>(qb, kb, vb, msk, cb);

        gemm_bf16_kernel<<<gH, blk, 0, stream>>>(cb, woT, bo + (size_t)l * HH, tmp, BS, HH, HH, 0);
        add_ln_kernel<<<BS / 8, blk, 0, stream>>>(x, tmp, ln1s + (size_t)l * HH, ln1b + (size_t)l * HH, xb);

        gemm_bf16_kernel<<<gF, blk, 0, stream>>>(xb, w1T, b1 + (size_t)l * FFF, fb, BS, FFF, HH, 1 | 2);
        gemm_bf16_kernel<<<gH, blk, 0, stream>>>(fb, w2T, b2 + (size_t)l * HH, tmp, BS, HH, FFF, 0);
        add_ln_kernel<<<BS / 8, blk, 0, stream>>>(x, tmp, ln2s + (size_t)l * HH, ln2b + (size_t)l * HH, xb);
    }

    // ---------- pool, dense2 (WMMA GEMM, M=32 guarded), logits ----------
    pool_kernel<<<BB, blk, 0, stream>>>(x, idx, poob);
    gemm_bf16_kernel<<<dim3(1, HH / 128), blk, 0, stream>>>(poob, d2Wt, d2b, out2, BB, HH, HH, 0);
    logits_kernel<<<BB, dim3(32, 3), 0, stream>>>(out2, wemb, pidx, (float*)d_out);
}
